// MultiGNN_22110491640455
// MI455X (gfx1250) — matrix operand order repaired and checked by reading, not compiled, not run on hardware
//
#include <hip/hip_runtime.h>
#include <hip/hip_bf16.h>

typedef __attribute__((ext_vector_type(16))) _Float16 v16h;
typedef __attribute__((ext_vector_type(8)))  _Float16 v8h;
typedef __attribute__((ext_vector_type(4)))  _Float16 v4h;
typedef __attribute__((ext_vector_type(8)))  float    v8f;

__device__ __forceinline__ v8f vzero8f() { v8f z = {}; return z; }

// D = A(16x32 f16) * B(32x16 f16) + C, f32 accum
__device__ __forceinline__ v8f wmma16(v16h a, v16h b, v8f c) {
  return __builtin_amdgcn_wmma_f32_16x16x32_f16(
      /*neg_a=*/false, a, /*neg_b=*/false, b,
      /*c_mod=*/(short)0, c, /*reuse_a=*/false, /*reuse_b=*/false);
}

// A-operand fetch from a row-major [M][ld] f16 matrix.
// Lane L: row = m, elements e: K = (L>>4)*16 + e
// -> two contiguous 8-half (16B) loads at colbase and colbase+8.
__device__ __forceinline__ v16h load_a16(const _Float16* p0, const _Float16* p1) {
  v8h lo = *(const v8h*)p0;
  v8h hi = *(const v8h*)(p0 + 8);
  v16h r;
#pragma unroll
  for (int i = 0; i < 8; ++i) { r[i] = lo[i]; r[i + 8] = hi[i]; }
  return r;
}

// wave index as an SGPR value -> scalar loop control / address math
__device__ __forceinline__ int wave_id() {
  return __builtin_amdgcn_readfirstlane(threadIdx.x >> 5);
}

// ---------------------------------------------------------------------------
// Kernel 1: fused 4-layer conv stack, one block per image.
// Activations live in LDS with padded stride MPAD so the D epilogue is a
// single unconditional ds_store_b128 per WMMA tile.
// ---------------------------------------------------------------------------
template <int CI, int CO, int HI, int WI, int HO, int WO, int K, int KPAD,
          int IN_STRIDE>
__device__ __forceinline__ void conv_layer(const _Float16* actIn,
                                           _Float16* actOut,
                                           const float* __restrict__ wgt,
                                           const float* __restrict__ bias,
                                           _Float16* sIm, _Float16* sWgt,
                                           float* sBias) {
  constexpr int M    = HO * WO;
  constexpr int MT   = (M + 15) >> 4;
  constexpr int MPAD = MT << 4;
  constexpr int NT   = CO >> 4;
  constexpr int KT   = KPAD >> 5;
  const int tid  = threadIdx.x;
  const int lane = tid & 31;
  const int wv   = wave_id();

  // weights -> Bt layout [CO][KPAD]; zero the K..KPAD pad (B pad MUST be 0)
  for (int idx = tid; idx < CO * KPAD; idx += 256) {
    int n  = idx / KPAD;
    int k  = idx - n * KPAD;
    int kk = k < K ? k : K - 1;                 // clamp load address
    float w = wgt[n * K + kk];
    sWgt[idx] = (k < K) ? (_Float16)w : (_Float16)0.0f;
  }
  if (tid < CO) sBias[tid] = bias[tid];

  // im2col [MPAD][KPAD], branchless with clamped indices.
  // Pad rows (m>=M) produce garbage D rows that land in the padded stride;
  // pad cols (k>=K) multiply against zeroed B.
  for (int idx = tid; idx < MPAD * KPAD; idx += 256) {
    int m  = idx / KPAD;
    int k  = idx - m * KPAD;
    int mm = m < M ? m : M - 1;
    int kk = k < K ? k : K - 1;
    int oh = mm / WO, ow = mm - oh * WO;
    int c  = kk / 9,  r9 = kk - c * 9;
    int kh = r9 / 3,  kw = r9 - kh * 3;
    sIm[idx] = actIn[c * IN_STRIDE + (oh + kh) * WI + (ow + kw)];
  }
  __syncthreads();

  constexpr int JOBS = MT * NT;
  for (int job = wv; job < JOBS; job += 8) {    // scalar loop (wv in SGPR)
    int mt = job / NT;
    int nt = job - mt * NT;
    v8f acc = vzero8f();
    const _Float16* arow = sIm  + ((mt << 4) + (lane & 15)) * KPAD + ((lane >> 4) << 4);
    const _Float16* brow = sWgt + ((nt << 4) + (lane & 15)) * KPAD + ((lane >> 4) << 4);
    // software-pipelined A loads
    v16h a = load_a16(arow, arow + 16);
#pragma unroll
    for (int kt = 0; kt < KT; ++kt) {
      v16h a_nxt;
      if (kt + 1 < KT)
        a_nxt = load_a16(arow + ((kt + 1) << 5), arow + ((kt + 1) << 5) + 16);
      v16h b = *(const v16h*)(brow + (kt << 5));
      acc = wmma16(a, b, acc);
      a = a_nxt;
    }
    // epilogue: bias+relu, pack 8 consecutive m per lane -> one b128 store
    const int n  = (nt << 4) + (lane & 15);
    const float bs = sBias[n];
    v8h o;
#pragma unroll
    for (int r = 0; r < 8; ++r) {
      float v = acc[r] + bs;
      o[r] = (_Float16)(v > 0.0f ? v : 0.0f);
    }
    *(v8h*)(actOut + n * MPAD + (mt << 4) + ((lane >> 4) << 3)) = o;
  }
  __syncthreads();
}

__global__ __launch_bounds__(256)
void conv_stack_kernel(const float* __restrict__ states,
                       const float* __restrict__ cw0, const float* __restrict__ cb0,
                       const float* __restrict__ cw1, const float* __restrict__ cb1,
                       const float* __restrict__ cw2, const float* __restrict__ cb2,
                       const float* __restrict__ cw3, const float* __restrict__ cb3,
                       _Float16* __restrict__ feat_out) {
  __shared__ __align__(32) _Float16 sAct0[4096];    // max padded act 32*128
  __shared__ __align__(32) _Float16 sAct1[4096];
  __shared__ __align__(32) _Float16 sIm[27648];     // max MPAD*KPAD = 96*288
  __shared__ __align__(32) _Float16 sWgt[5120];     // max CO*KPAD = 32*160
  __shared__ float sBias[32];

  const int img = blockIdx.x;
  const int tid = threadIdx.x;

  // load image 8*15*15 f32 -> f16 (dense stride 225), vectorized x4
  const float4* src4 = (const float4*)(states + (size_t)img * 1800);
  for (int i = tid; i < 450; i += 256) {
    float4 f = src4[i];
    v4h h;
    h[0] = (_Float16)f.x; h[1] = (_Float16)f.y;
    h[2] = (_Float16)f.z; h[3] = (_Float16)f.w;
    *(v4h*)(sAct0 + i * 4) = h;
  }
  __syncthreads();

  // IN_STRIDE chain: 225 (dense) -> 176 -> 128 -> 96 (padded MPADs)
  conv_layer< 8, 16, 15, 15, 13, 13,  72,  96, 225>(sAct0, sAct1, cw0, cb0, sIm, sWgt, sBias);
  conv_layer<16, 32, 13, 13, 11, 11, 144, 160, 176>(sAct1, sAct0, cw1, cb1, sIm, sWgt, sBias);
  conv_layer<32, 16, 11, 11,  9,  9, 288, 288, 128>(sAct0, sAct1, cw2, cb2, sIm, sWgt, sBias);
  conv_layer<16, 32,  9,  9,  7,  7, 144, 160,  96>(sAct1, sAct0, cw3, cb3, sIm, sWgt, sBias);

  // compact padded [32][64] -> exact flatten order [32][49]
  _Float16* dst = feat_out + (size_t)img * 1568;
  for (int i = tid; i < 1568; i += 256) {
    int c = i / 49, m = i - c * 49;
    dst[i] = sAct0[c * 64 + m];
  }
}

// ---------------------------------------------------------------------------
// Kernel 2: fused 3-layer MLP, 64 output rows per block, 8 waves tile 64x128.
// GEMM0 double-buffers its 32x128 weight slab and prefetches the next global
// A fragment across the barrier.
// ---------------------------------------------------------------------------
__global__ __launch_bounds__(256)
void mlp_kernel(const _Float16* __restrict__ featA,   // [16384][1568] f16
                const float* __restrict__ mw0, const float* __restrict__ mb0,
                const float* __restrict__ mw1, const float* __restrict__ mb1,
                const float* __restrict__ mw2, const float* __restrict__ mb2,
                _Float16* __restrict__ featB) {       // [16384][128] f16
  __shared__ __align__(32) _Float16 ldsW[128 * 128]; // Bt weights (32KB)
  __shared__ __align__(32) _Float16 ldsX[64 * 128];  // activations (16KB)

  const int tid  = threadIdx.x;
  const int lane = tid & 31;
  const int wv   = wave_id();
  const int mt   = wv & 3;        // 16-row strip within block
  const int nh   = wv >> 2;       // which 64-col half
  const int m0   = blockIdx.x * 64;

  v8f acc[4];
#pragma unroll
  for (int nt = 0; nt < 4; ++nt) acc[nt] = vzero8f();

  // ---- GEMM0: [64 x 1568] @ [1568 x 128], K-steps of 32, double-buffered ----
  const _Float16* abase =
      featA + (size_t)(m0 + (mt << 4) + (lane & 15)) * 1568;
  auto stage0 = [&](int kt, _Float16* buf) {
    for (int idx = tid; idx < 2048; idx += 256) {       // float2-coalesced
      int n2 = (idx & 63) << 1, k = idx >> 6;
      float2 w = *(const float2*)&mw0[(size_t)(kt * 32 + k) * 128 + n2];
      buf[n2 * 32 + k]       = (_Float16)w.x;
      buf[(n2 + 1) * 32 + k] = (_Float16)w.y;
    }
  };
  _Float16* buf0 = ldsW;
  _Float16* buf1 = ldsW + 32 * 128;
  stage0(0, buf0);
  v16h a = load_a16(abase + ((lane >> 4) << 4),
                    abase + ((lane >> 4) << 4) + 16);
  __syncthreads();
  for (int kt = 0; kt < 49; ++kt) {
    _Float16* cur = (kt & 1) ? buf1 : buf0;
    if (kt < 48) {
      stage0(kt + 1, (kt & 1) ? buf0 : buf1);
      const _Float16* ap = abase + (kt + 1) * 32 + ((lane >> 4) << 4);
      v16h a_nxt = load_a16(ap, ap + 16);
#pragma unroll
      for (int nt = 0; nt < 4; ++nt) {
        const _Float16* bp =
            cur + ((nh << 6) + (nt << 4) + (lane & 15)) * 32 + ((lane >> 4) << 4);
        acc[nt] = wmma16(a, *(const v16h*)bp, acc[nt]);
      }
      a = a_nxt;
    } else {
#pragma unroll
      for (int nt = 0; nt < 4; ++nt) {
        const _Float16* bp =
            cur + ((nh << 6) + (nt << 4) + (lane & 15)) * 32 + ((lane >> 4) << 4);
        acc[nt] = wmma16(a, *(const v16h*)bp, acc[nt]);
      }
    }
    __syncthreads();
  }
  // relu + bias -> ldsX in A layout [m][128]
#pragma unroll
  for (int nt = 0; nt < 4; ++nt) {
    int n = (nh << 6) + (nt << 4) + (lane & 15);
    float bs = mb0[n];
#pragma unroll
    for (int r = 0; r < 8; ++r) {
      int ml = (mt << 4) + r + ((lane >> 4) << 3);
      float v = acc[nt][r] + bs;
      ldsX[ml * 128 + n] = (_Float16)(v > 0.0f ? v : 0.0f);
    }
    acc[nt] = vzero8f();
  }
  __syncthreads();

  // ---- GEMM1: [64 x 128] @ [128 x 128] ----
  for (int idx = tid; idx < 8192; idx += 256) {
    int n2 = (idx & 63) << 1, k = idx >> 6;
    float2 w = *(const float2*)&mw1[(size_t)k * 128 + n2];
    ldsW[n2 * 128 + k]       = (_Float16)w.x;
    ldsW[(n2 + 1) * 128 + k] = (_Float16)w.y;
  }
  __syncthreads();
  {
    const _Float16* ax = ldsX + ((mt << 4) + (lane & 15)) * 128 + ((lane >> 4) << 4);
    v16h a1 = load_a16(ax, ax + 16);
#pragma unroll
    for (int kt = 0; kt < 4; ++kt) {
      v16h a_nxt;
      if (kt < 3) a_nxt = load_a16(ax + ((kt + 1) << 5), ax + ((kt + 1) << 5) + 16);
#pragma unroll
      for (int nt = 0; nt < 4; ++nt) {
        const _Float16* bp =
            ldsW + ((nh << 6) + (nt << 4) + (lane & 15)) * 128 + (kt << 5) + ((lane >> 4) << 4);
        acc[nt] = wmma16(a1, *(const v16h*)bp, acc[nt]);
      }
      a1 = a_nxt;
    }
  }
  __syncthreads();
#pragma unroll
  for (int nt = 0; nt < 4; ++nt) {
    int n = (nh << 6) + (nt << 4) + (lane & 15);
    float bs = mb1[n];
#pragma unroll
    for (int r = 0; r < 8; ++r) {
      int ml = (mt << 4) + r + ((lane >> 4) << 3);
      float v = acc[nt][r] + bs;
      ldsX[ml * 128 + n] = (_Float16)(v > 0.0f ? v : 0.0f);
    }
    acc[nt] = vzero8f();
  }
  __syncthreads();

  // ---- GEMM2 (linear): [64 x 128] @ [128 x 128] ----
  for (int idx = tid; idx < 8192; idx += 256) {
    int n2 = (idx & 63) << 1, k = idx >> 6;
    float2 w = *(const float2*)&mw2[(size_t)k * 128 + n2];
    ldsW[n2 * 128 + k]       = (_Float16)w.x;
    ldsW[(n2 + 1) * 128 + k] = (_Float16)w.y;
  }
  __syncthreads();
  {
    const _Float16* ax = ldsX + ((mt << 4) + (lane & 15)) * 128 + ((lane >> 4) << 4);
    v16h a2 = load_a16(ax, ax + 16);
#pragma unroll
    for (int kt = 0; kt < 4; ++kt) {
      v16h a_nxt;
      if (kt < 3) a_nxt = load_a16(ax + ((kt + 1) << 5), ax + ((kt + 1) << 5) + 16);
#pragma unroll
      for (int nt = 0; nt < 4; ++nt) {
        const _Float16* bp =
            ldsW + ((nh << 6) + (nt << 4) + (lane & 15)) * 128 + (kt << 5) + ((lane >> 4) << 4);
        acc[nt] = wmma16(a2, *(const v16h*)bp, acc[nt]);
      }
      a2 = a_nxt;
    }
  }
#pragma unroll
  for (int nt = 0; nt < 4; ++nt) {
    int n = (nh << 6) + (nt << 4) + (lane & 15);
    float bs = mb2[n];
#pragma unroll
    for (int r = 0; r < 8; ++r) {
      int ml = (mt << 4) + r + ((lane >> 4) << 3);
      featB[(size_t)(m0 + ml) * 128 + n] = (_Float16)(acc[nt][r] + bs);
    }
  }
}

// ---------------------------------------------------------------------------
// Kernel 3: GCN — normalized adjacency aggregation, one block per batch b
// ---------------------------------------------------------------------------
__global__ __launch_bounds__(256)
void gcn_kernel(const float* __restrict__ adj,     // [256][64][64]
                const _Float16* __restrict__ feat, // [16384][128] f16
                const float* __restrict__ gw, const float* __restrict__ gb,
                float* __restrict__ out) {         // [16384][128] f32
  __shared__ __align__(32) _Float16 sNormT[64 * 64];   // [j][i]  (8KB)
  __shared__ __align__(32) _Float16 sGwT[128 * 128];   // Bt of gw (32KB)
  __shared__ __align__(32) _Float16 sXwT[128 * 64];    // [e][i]  (16KB)
  __shared__ float sDinv[64];

  const int b    = blockIdx.x;
  const float* A = adj + (size_t)b * 4096;
  const int tid  = threadIdx.x;
  const int lane = tid & 31;
  const int wv   = wave_id();
  const int mt   = wv & 3;
  const int nh   = wv >> 2;

  if (tid < 64) {
    float d = 0.0f;
    for (int i = 0; i < 64; ++i) d += A[i * 64 + tid];   // deg[j] = col sum
    sDinv[tid] = d > 0.0f ? rsqrtf(d) : 0.0f;
  }
  __syncthreads();
  for (int idx = tid; idx < 4096; idx += 256) {
    int j = idx >> 6, i = idx & 63;
    sNormT[idx] = (_Float16)(sDinv[i] * A[i * 64 + j] * sDinv[j]);
  }
  for (int idx = tid; idx < 8192; idx += 256) {
    int n2 = (idx & 63) << 1, k = idx >> 6;
    float2 w = *(const float2*)&gw[(size_t)k * 128 + n2];
    sGwT[n2 * 128 + k]       = (_Float16)w.x;
    sGwT[(n2 + 1) * 128 + k] = (_Float16)w.y;
  }
  __syncthreads();

  // xw = feats_b (64x128) @ gw (128x128)
  v8f acc[4];
#pragma unroll
  for (int nt = 0; nt < 4; ++nt) acc[nt] = vzero8f();

  const _Float16* abase =
      feat + (size_t)(b * 64 + (mt << 4) + (lane & 15)) * 128;
  {
    v16h a = load_a16(abase + ((lane >> 4) << 4),
                      abase + ((lane >> 4) << 4) + 16);
#pragma unroll
    for (int kt = 0; kt < 4; ++kt) {
      v16h a_nxt;
      if (kt < 3) {
        const _Float16* ap = abase + ((kt + 1) << 5) + ((lane >> 4) << 4);
        a_nxt = load_a16(ap, ap + 16);
      }
#pragma unroll
      for (int nt = 0; nt < 4; ++nt) {
        const _Float16* bp =
            sGwT + ((nh << 6) + (nt << 4) + (lane & 15)) * 128 + (kt << 5) + ((lane >> 4) << 4);
        acc[nt] = wmma16(a, *(const v16h*)bp, acc[nt]);
      }
      a = a_nxt;
    }
  }
  // store xw transposed [e][i] == B-operand layout for the aggregation
#pragma unroll
  for (int nt = 0; nt < 4; ++nt) {
    int e = (nh << 6) + (nt << 4) + (lane & 15);
#pragma unroll
    for (int r = 0; r < 8; ++r) {
      int il = (mt << 4) + r + ((lane >> 4) << 3);
      sXwT[e * 64 + il] = (_Float16)acc[nt][r];
    }
    acc[nt] = vzero8f();
  }
  __syncthreads();

  // out_b (64x128) = normT (64x64) @ xw (64x128), then + gb
  {
    const _Float16* arow = sNormT + ((mt << 4) + (lane & 15)) * 64 + ((lane >> 4) << 4);
    v16h a = load_a16(arow, arow + 16);
#pragma unroll
    for (int kt = 0; kt < 2; ++kt) {
      v16h a_nxt;
      if (kt < 1) a_nxt = load_a16(arow + 32, arow + 48);
#pragma unroll
      for (int nt = 0; nt < 4; ++nt) {
        const _Float16* bp =
            sXwT + ((nh << 6) + (nt << 4) + (lane & 15)) * 64 + (kt << 5) + ((lane >> 4) << 4);
        acc[nt] = wmma16(a, *(const v16h*)bp, acc[nt]);
      }
      a = a_nxt;
    }
  }
#pragma unroll
  for (int nt = 0; nt < 4; ++nt) {
    int e = (nh << 6) + (nt << 4) + (lane & 15);
    float bs = gb[e];
#pragma unroll
    for (int r = 0; r < 8; ++r) {
      int j = (mt << 4) + r + ((lane >> 4) << 3);
      out[((size_t)(b * 64 + j)) * 128 + e] = acc[nt][r] + bs;
    }
  }
}

// ---------------------------------------------------------------------------
extern "C" void kernel_launch(void* const* d_in, const int* in_sizes, int n_in,
                              void* d_out, int out_size, void* d_ws, size_t ws_size,
                              hipStream_t stream) {
  const float* states = (const float*)d_in[0];
  const float* adj    = (const float*)d_in[1];
  const float* cw0 = (const float*)d_in[2];
  const float* cb0 = (const float*)d_in[3];
  const float* cw1 = (const float*)d_in[4];
  const float* cb1 = (const float*)d_in[5];
  const float* cw2 = (const float*)d_in[6];
  const float* cb2 = (const float*)d_in[7];
  const float* cw3 = (const float*)d_in[8];
  const float* cb3 = (const float*)d_in[9];
  const float* mw0 = (const float*)d_in[10];
  const float* mb0 = (const float*)d_in[11];
  const float* mw1 = (const float*)d_in[12];
  const float* mb1 = (const float*)d_in[13];
  const float* mw2 = (const float*)d_in[14];
  const float* mb2 = (const float*)d_in[15];
  const float* gw  = (const float*)d_in[16];
  const float* gb  = (const float*)d_in[17];
  float* out = (float*)d_out;

  _Float16* wsConv = (_Float16*)d_ws;                       // 16384*1568 f16
  _Float16* wsFeat = wsConv + (size_t)16384 * 1568;         // 16384*128  f16

  conv_stack_kernel<<<16384, 256, 0, stream>>>(states, cw0, cb0, cw1, cb1,
                                               cw2, cb2, cw3, cb3, wsConv);
  mlp_kernel<<<256, 256, 0, stream>>>(wsConv, mw0, mb0, mw1, mb1, mw2, mb2,
                                      wsFeat);
  gcn_kernel<<<256, 256, 0, stream>>>(adj, wsFeat, gw, gb, out);
}
